// EdgePromptplus_18537078849988
// MI455X (gfx1250) — matrix operand order repaired and checked by reading, not compiled, not run on hardware
//
#include <hip/hip_runtime.h>

typedef __attribute__((ext_vector_type(16))) _Float16 v16h;
typedef __attribute__((ext_vector_type(8)))  float    v8f;

#define NN 50000
#define EE 400000
#define DD 128
#define AA 64
#define MROWS (NN + EE)          // 450000
#define TILES (MROWS / 16)       // 28125 (exact)
#define WAVES 8
#define THREADS 256
#define SW_BYTES 32768           // W as 32 B-fragments of 1KB (f16)
#define SA_BYTES 16384           // anchor as 16 B-fragments of 1KB (f16)
#define SC_BYTES (WAVES * 2048)  // per-wave coef tile in A-fragment order

__global__ __launch_bounds__(THREADS)
void edgeprompt_kernel(const float* __restrict__ x,
                       const long long* __restrict__ ei,
                       const float* __restrict__ Wg,
                       const float* __restrict__ bg,
                       const float* __restrict__ Ag,
                       float* __restrict__ out)
{
    __shared__ __align__(32) char smem[SW_BYTES + SA_BYTES + SC_BYTES];

    const int tid  = threadIdx.x;
    const int lane = tid & 31;
    const int wave = tid >> 5;
    const int q    = lane & 15;   // column-in-tile / row-in-tile index
    const int hi   = lane >> 4;   // 0: lanes 0-15, 1: lanes 16-31

    // ---- stage W (A x 2D) f32 -> f16, fragment-major for GEMM1 B operands ----
    // B[k][n] = W[n][k]; fragment(kc,nt): lane ls holds 16 contiguous f16.
    for (int i = tid; i < AA * 2 * DD; i += THREADS) {
        int n  = i >> 8;  int k = i & 255;
        int kc = k >> 5;  int nt = n >> 4;
        int ls  = (n & 15) | (((k >> 3) & 1) << 4);
        int sub = (k & 7) | (((k >> 4) & 1) << 3);
        *(_Float16*)(smem + ((kc << 2) + nt) * 1024 + ls * 32 + sub * 2) =
            (_Float16)Wg[i];
    }
    // ---- stage anchor (A x D) f32 -> f16, fragment-major for GEMM2 B operands ----
    // B[k=a][n=d] = anchor[a][d]
    for (int i = tid; i < AA * DD; i += THREADS) {
        int a   = i >> 7;  int d = i & 127;
        int kc2 = a >> 5;  int r32 = a & 31;
        int ls  = (d & 15) | (((r32 >> 3) & 1) << 4);
        int j   = (r32 & 7) | ((r32 >> 4) << 3); int nt2 = d >> 4;
        *(_Float16*)(smem + SW_BYTES + ((kc2 << 3) + nt2) * 1024 + ls * 32 + j * 2) =
            (_Float16)Ag[i];
    }
    __syncthreads();

    // bias per lane: column nt*16+q of each N-tile
    float bv[4];
#pragma unroll
    for (int nt = 0; nt < 4; ++nt) bv[nt] = bg[nt * 16 + q];

    char* coefBase = smem + SW_BYTES + SA_BYTES + wave * 2048;

    const int gw      = blockIdx.x * WAVES + wave;
    const int gstride = gridDim.x * WAVES;

    for (int tile = gw; tile < TILES; tile += gstride) {
        const int row = tile * 16 + q;
        long long s, d;
        if (row < EE) { s = ei[row]; d = ei[EE + row]; }
        else          { s = d = row - EE; }
        const float* xs = x + (size_t)s * DD;
        const float* xd = x + (size_t)d * DD;

        // ---- GEMM1: logits(16x64) = combined(16x256) @ W^T, acc init = bias ----
        v8f acc[4];
#pragma unroll
        for (int nt = 0; nt < 4; ++nt)
#pragma unroll
            for (int j = 0; j < 8; ++j) acc[nt][j] = bv[nt];

#pragma unroll
        for (int kc = 0; kc < 8; ++kc) {
            const float* p  = (kc < 4) ? (xs + kc * 32) : (xd + (kc - 4) * 32);
            const float* s0 = p + hi * 8;
            float4 f0 = *(const float4*)(s0);
            float4 f1 = *(const float4*)(s0 + 4);
            float4 f2 = *(const float4*)(s0 + 16);
            float4 f3 = *(const float4*)(s0 + 20);
            v16h a;
            a[0]  = (_Float16)f0.x; a[1]  = (_Float16)f0.y;
            a[2]  = (_Float16)f0.z; a[3]  = (_Float16)f0.w;
            a[4]  = (_Float16)f1.x; a[5]  = (_Float16)f1.y;
            a[6]  = (_Float16)f1.z; a[7]  = (_Float16)f1.w;
            a[8]  = (_Float16)f2.x; a[9]  = (_Float16)f2.y;
            a[10] = (_Float16)f2.z; a[11] = (_Float16)f2.w;
            a[12] = (_Float16)f3.x; a[13] = (_Float16)f3.y;
            a[14] = (_Float16)f3.z; a[15] = (_Float16)f3.w;
#pragma unroll
            for (int nt = 0; nt < 4; ++nt) {
                v16h bf = *(const v16h*)(smem + ((kc << 2) + nt) * 1024 + lane * 32);
                acc[nt] = __builtin_amdgcn_wmma_f32_16x16x32_f16(
                    false, a, false, bf, (short)0, acc[nt], false, false);
            }
        }

        // ---- leaky_relu + row softmax (rows in VGPR index, cols in lanes) ----
        // store coef f16 into per-wave LDS already in GEMM2 A-fragment order
#pragma unroll
        for (int j = 0; j < 8; ++j) {
            float l0 = acc[0][j], l1 = acc[1][j], l2 = acc[2][j], l3 = acc[3][j];
            l0 = l0 > 0.f ? l0 : 0.01f * l0;
            l1 = l1 > 0.f ? l1 : 0.01f * l1;
            l2 = l2 > 0.f ? l2 : 0.01f * l2;
            l3 = l3 > 0.f ? l3 : 0.01f * l3;
            float mx = fmaxf(fmaxf(l0, l1), fmaxf(l2, l3));
            mx = fmaxf(mx, __shfl_xor(mx, 1, 32));
            mx = fmaxf(mx, __shfl_xor(mx, 2, 32));
            mx = fmaxf(mx, __shfl_xor(mx, 4, 32));
            mx = fmaxf(mx, __shfl_xor(mx, 8, 32));
            float e0 = __expf(l0 - mx), e1 = __expf(l1 - mx);
            float e2 = __expf(l2 - mx), e3 = __expf(l3 - mx);
            float sm = e0 + e1 + e2 + e3;
            sm += __shfl_xor(sm, 1, 32);
            sm += __shfl_xor(sm, 2, 32);
            sm += __shfl_xor(sm, 4, 32);
            sm += __shfl_xor(sm, 8, 32);
            float inv = 1.0f / sm;
            float ce[4] = { e0 * inv, e1 * inv, e2 * inv, e3 * inv };
            int r = j + (hi << 3);               // tile row this value belongs to
#pragma unroll
            for (int t = 0; t < 4; ++t) {
                int rel  = ((t & 1) << 4) | q;   // K offset within 32-chunk
                int half = (rel >> 3) & 1;
                int pos  = (rel & 7) | ((rel >> 4) << 3);
                int wl   = r + (half << 4);      // lane that will consume it
                *(_Float16*)(coefBase + ((t >> 1) << 10) + wl * 32 + pos * 2) =
                    (_Float16)ce[t];
            }
        }

        // ---- GEMM2: out(16x128) = coef(16x64) @ anchor(64x128) ----
        v8f acc2[8];
#pragma unroll
        for (int nt2 = 0; nt2 < 8; ++nt2)
#pragma unroll
            for (int j = 0; j < 8; ++j) acc2[nt2][j] = 0.f;

#pragma unroll
        for (int kc2 = 0; kc2 < 2; ++kc2) {
            v16h ca = *(const v16h*)(coefBase + (kc2 << 10) + lane * 32);
#pragma unroll
            for (int nt2 = 0; nt2 < 8; ++nt2) {
                v16h bb = *(const v16h*)(smem + SW_BYTES +
                                         ((kc2 << 3) + nt2) * 1024 + lane * 32);
                acc2[nt2] = __builtin_amdgcn_wmma_f32_16x16x32_f16(
                    false, ca, false, bb, (short)0, acc2[nt2], false, false);
            }
        }

        // ---- store 16x128 f32 tile (coalesced 64B per 16-lane group) ----
        float* orow = out + (size_t)tile * (16 * DD);
#pragma unroll
        for (int nt2 = 0; nt2 < 8; ++nt2)
#pragma unroll
            for (int j = 0; j < 8; ++j)
                orow[(j + (hi << 3)) * DD + nt2 * 16 + q] = acc2[nt2][j];
    }
}

extern "C" void kernel_launch(void* const* d_in, const int* in_sizes, int n_in,
                              void* d_out, int out_size, void* d_ws, size_t ws_size,
                              hipStream_t stream) {
    (void)in_sizes; (void)n_in; (void)out_size; (void)d_ws; (void)ws_size;
    const float*     x      = (const float*)d_in[0];
    const long long* ei     = (const long long*)d_in[1];  // int64 edge_index (2,E) flat
    // d_in[2] = layer scalar (unused)
    const float*     W      = (const float*)d_in[3];
    const float*     b      = (const float*)d_in[4];
    const float*     anchor = (const float*)d_in[5];
    float*           out    = (float*)d_out;

    dim3 grid(704), block(THREADS);
    edgeprompt_kernel<<<grid, block, 0, stream>>>(x, ei, W, b, anchor, out);
}
